// MultiqueryAttention_40638980555072
// MI455X (gfx1250) — matrix operand order repaired and checked by reading, not compiled, not run on hardware
//
#include <hip/hip_runtime.h>
#include <hip/hip_bf16.h>
#include <stdint.h>

// ---------------- problem constants ----------------
#define S_LEN   2048
#define NHEADS  16
#define DHEAD   64
#define DMODEL  1024
#define DQKV    1152            // DMODEL + 2*DHEAD
#define BATCH   2
#define NROWS   (BATCH * S_LEN) // 4096

typedef __bf16 bf16_t;
typedef __attribute__((ext_vector_type(16))) __bf16    v16bf;
typedef __attribute__((ext_vector_type(8)))  __bf16    v8bf;
typedef __attribute__((ext_vector_type(8)))  float     v8f;
typedef __attribute__((ext_vector_type(4)))  uint32_t  u32x4;
typedef __attribute__((ext_vector_type(8)))  uint32_t  u32x8;

// ---------------- WMMA fragment loaders (per CDNA5 ISA 7.12.2 layouts) ------
// A (16xK, row-major, K-chunk of 32):
//   lane L<16 : row L, elems {c..c+7, c+16..c+23}
//   lane L>=16: row L-16, elems {c+8..c+15, c+24..c+31}
static __device__ __forceinline__ v16bf load_a_bf16(const bf16_t* base, int ld,
                                                    int row0, int col0, int lane) {
  int r = row0 + (lane & 15);
  const bf16_t* p = base + (size_t)r * ld + col0 + ((lane >> 4) << 3);
  v8bf lo = *(const v8bf*)p;
  v8bf hi = *(const v8bf*)(p + 16);
  v16bf a;
#pragma unroll
  for (int i = 0; i < 8; ++i) { a[i] = lo[i]; a[i + 8] = hi[i]; }
  return a;
}

// Same A-fragment but source is f32 (attn activations), cast on the fly.
static __device__ __forceinline__ v16bf load_a_f32(const float* base, int ld,
                                                   int row0, int col0, int lane) {
  int r = row0 + (lane & 15);
  const float* p = base + (size_t)r * ld + col0 + ((lane >> 4) << 3);
  v16bf a;
#pragma unroll
  for (int i = 0; i < 8; ++i) { a[i] = (bf16_t)p[i]; a[i + 8] = (bf16_t)p[16 + i]; }
  return a;
}

// B (Kx16) sourced from a transposed (N x K row-major) operand (global or LDS):
//   lane L<16 : col L,  k = {c..c+7, c+16..c+23}
//   lane L>=16: col L-16, k = {c+8..c+15, c+24..c+31}
static __device__ __forceinline__ v16bf load_b_bf16(const bf16_t* baseT, int ld,
                                                    int col0, int c, int lane) {
  int n = col0 + (lane & 15);
  const bf16_t* p = baseT + (size_t)n * ld + c + ((lane >> 4) << 3);
  v8bf lo = *(const v8bf*)p;
  v8bf hi = *(const v8bf*)(p + 16);
  v16bf b;
#pragma unroll
  for (int i = 0; i < 8; ++i) { b[i] = lo[i]; b[i + 8] = hi[i]; }
  return b;
}

static __device__ __forceinline__ v8f wmma_bf16(v16bf a, v16bf b, v8f c) {
  return __builtin_amdgcn_wmma_f32_16x16x32_bf16(false, a, false, b, (short)0, c,
                                                 false, false);
}

// ---------------- Tensor Data Mover: 2D tile (16-bit elements) --------------
// Builds D# group0 (count/lds_addr/global_addr/type=2) and group1
// (data_size=2B, tensor dims, tile dims, dim0 stride) per ISA 8.3/8.4 and
// issues tensor_load_to_lds with the two SGPR descriptor groups.
static __device__ __forceinline__ uint32_t lds_off32(const void* p) {
  return (uint32_t)(uintptr_t)p;   // low 32 bits of LDS aperture addr = offset
}

static __device__ __forceinline__ void tdm_load_2d_bf16(uint32_t lds_addr,
                                                        const bf16_t* gptr,
                                                        uint32_t tensor_d0,
                                                        uint32_t tensor_d1,
                                                        uint32_t tile_d0,
                                                        uint32_t tile_d1,
                                                        uint32_t stride0) {
  uint64_t ga = (uint64_t)(uintptr_t)gptr;
  u32x4 g0;
  g0[0] = 1u;                                                // count=1, user D#
  g0[1] = lds_addr;                                          // lds_addr[31:0]
  g0[2] = (uint32_t)ga;                                      // global_addr lo
  g0[3] = ((uint32_t)(ga >> 32) & 0x01FFFFFFu) | 0x80000000u;// addr hi | type=2
  u32x8 g1;
  g1[0] = 0x00010000u;                  // workgroup_mask=0, data_size=1 (2B)
  g1[1] = tensor_d0 << 16;              // tensor_dim0[15:0]  @ bits 63:48
  g1[2] = (tensor_d0 >> 16) | (tensor_d1 << 16);  // dim0 hi | dim1 lo
  g1[3] = (tensor_d1 >> 16) | (tile_d0 << 16);    // dim1 hi | tile_dim0
  g1[4] = tile_d1 & 0xFFFFu;            // tile_dim1 (tile_dim2 = 0 -> 2D)
  g1[5] = stride0;                      // tensor_dim0_stride[31:0]
  g1[6] = 0u;
  g1[7] = 0u;
  asm volatile("tensor_load_to_lds %0, %1" :: "s"(g0), "s"(g1) : "memory");
}

// ---------------- elementwise prep kernels ----------------
__global__ void cast_f32_bf16(const float* __restrict__ in, bf16_t* __restrict__ out, int n) {
  int i = blockIdx.x * blockDim.x + threadIdx.x;
  if (i < n) out[i] = (bf16_t)in[i];
}

// out[n*K + k] = bf16(in[k*N + n])   (K x N row-major  ->  N x K row-major)
__global__ void transpose_cast(const float* __restrict__ in, bf16_t* __restrict__ out,
                               int K, int N) {
  int i = blockIdx.x * blockDim.x + threadIdx.x;
  if (i < K * N) {
    int k = i / N, n = i - k * N;
    out[(size_t)n * K + k] = (bf16_t)in[i];
  }
}

// ---------------- QKV projection: (4096x1024)@(1024x1152)+b -----------------
// one wave per 16x128 output tile: one A-fragment feeds 8 independent WMMAs
__global__ void qkv_gemm(const bf16_t* __restrict__ xb, const bf16_t* __restrict__ wT,
                         const float* __restrict__ bias, float* __restrict__ qkv) {
  int lane = threadIdx.x;
  int row0 = blockIdx.x * 16;
  int col0 = blockIdx.y * 128;
  v8f acc[8] = {};
  for (int c = 0; c < DMODEL; c += 32) {
    __builtin_prefetch(xb + (size_t)row0 * DMODEL + c + 64, 0, 1);
    v16bf a = load_a_bf16(xb, DMODEL, row0, c, lane);
#pragma unroll
    for (int t = 0; t < 8; ++t)
      acc[t] = wmma_bf16(a, load_b_bf16(wT, DMODEL, col0 + 16 * t, c, lane), acc[t]);
  }
  int nn = lane & 15;
#pragma unroll
  for (int t = 0; t < 8; ++t) {
#pragma unroll
    for (int r = 0; r < 8; ++r) {
      int mm = r + ((lane >> 4) << 3);
      size_t row = (size_t)(row0 + mm);
      qkv[row * DQKV + col0 + 16 * t + nn] = acc[t][r] + bias[col0 + 16 * t + nn];
    }
  }
}

// ---------------- RoPE + split to bf16 q, k, V^T ----------------------------
__global__ void rope_split(const float* __restrict__ qkv, bf16_t* __restrict__ qb,
                           bf16_t* __restrict__ kb, bf16_t* __restrict__ vT) {
  int row = blockIdx.x;            // 0..4095 = b*S + s
  int t = threadIdx.x;             // 0..255
  int s = row & (S_LEN - 1);
  int b = row >> 11;
  const float* src = qkv + (size_t)row * DQKV;
  const float C = 0.28782313662425572f;           // ln(10000)/32
#pragma unroll
  for (int pp = 0; pp < 2; ++pp) {
    int p = t + pp * 256;          // q pair index 0..511 ; head = p>>5, i = p&31
    int i = p & 31;
    float ang = (float)s * __expf(-(float)i * C);
    float cs = __cosf(ang), sn = __sinf(ang);
    float x0 = src[2 * p], x1 = src[2 * p + 1];
    qb[(size_t)row * DMODEL + 2 * p]     = (bf16_t)(x0 * cs - x1 * sn);
    qb[(size_t)row * DMODEL + 2 * p + 1] = (bf16_t)(x0 * sn + x1 * cs);
  }
  if (t < 32) {                    // shared K head, 32 pairs
    float ang = (float)s * __expf(-(float)t * C);
    float cs = __cosf(ang), sn = __sinf(ang);
    float x0 = src[DMODEL + 2 * t], x1 = src[DMODEL + 2 * t + 1];
    kb[(size_t)row * DHEAD + 2 * t]     = (bf16_t)(x0 * cs - x1 * sn);
    kb[(size_t)row * DHEAD + 2 * t + 1] = (bf16_t)(x0 * sn + x1 * cs);
  }
  if (t < 64) {                    // V transposed: (b, d, s)
    vT[((size_t)b * DHEAD + t) * S_LEN + s] = (bf16_t)src[DMODEL + DHEAD + t];
  }
}

// ---------------- flash multi-query attention -------------------------------
// One wave per (batch, head, 16-row q tile). K (32x64) and V^T (64x32) tiles
// are staged into LDS by the Tensor Data Mover, double-buffered: block i+1's
// two TDM descriptors are issued before computing block i, then
// s_wait_tensorcnt 2 leaves them in flight while WMMAs run from LDS.
__global__ void mqa_flash(const bf16_t* __restrict__ qb, const bf16_t* __restrict__ kb,
                          const bf16_t* __restrict__ vT, float* __restrict__ attn) {
  __shared__ __align__(128) bf16_t Klds[2][32 * DHEAD];   // keys x d
  __shared__ __align__(128) bf16_t Vlds[2][DHEAD * 32];   // d x keys (V^T tile)
  __shared__ __align__(128) float  Plds[16 * 32];
  int lane = threadIdx.x;
  int q0 = blockIdx.x * 16;
  int h  = blockIdx.y;
  int b  = blockIdx.z;
  const bf16_t* qB = qb + (size_t)b * S_LEN * DMODEL;
  const bf16_t* kB = kb + (size_t)b * S_LEN * DHEAD;
  const bf16_t* vB = vT + (size_t)b * DHEAD * S_LEN;

  v16bf qa0 = load_a_bf16(qB, DMODEL, q0, h * DHEAD + 0,  lane);
  v16bf qa1 = load_a_bf16(qB, DMODEL, q0, h * DHEAD + 32, lane);

  float mrow[8], lrow[8];
#pragma unroll
  for (int r = 0; r < 8; ++r) { mrow[r] = -3.0e38f; lrow[r] = 0.f; }
  v8f o0 = {}, o1 = {}, o2 = {}, o3 = {};

  int nn = lane & 15;
  int mg_base = q0 + ((lane >> 4) << 3);
  int nblk = (q0 + 47) >> 5;       // causal: cover keys 0..q0+15

  // prologue: stage block 0
  tdm_load_2d_bf16(lds_off32(Klds[0]), kB, DHEAD, S_LEN, DHEAD, 32, DHEAD);
  tdm_load_2d_bf16(lds_off32(Vlds[0]), vB, S_LEN, DHEAD, 32, DHEAD, S_LEN);

  for (int blk = 0; blk < nblk; ++blk) {
    int kv0 = blk * 32;
    int cur = blk & 1;
    if (blk + 1 < nblk) {          // prefetch next block into other buffer
      tdm_load_2d_bf16(lds_off32(Klds[cur ^ 1]), kB + (size_t)(kv0 + 32) * DHEAD,
                       DHEAD, S_LEN, DHEAD, 32, DHEAD);
      tdm_load_2d_bf16(lds_off32(Vlds[cur ^ 1]), vB + (kv0 + 32),
                       S_LEN, DHEAD, 32, DHEAD, S_LEN);
      __builtin_amdgcn_s_wait_tensorcnt(2);   // current block's 2 TDMs done
    } else {
      __builtin_amdgcn_s_wait_tensorcnt(0);
    }
    const bf16_t* Kbuf = Klds[cur];
    const bf16_t* Vbuf = Vlds[cur];

    // scores S = Q K^T for 16 q-rows x 32 keys (two 16x16 tiles, K-depth 64)
    v8f s0 = {}, s1 = {};
    s0 = wmma_bf16(qa0, load_b_bf16(Kbuf, DHEAD, 0,  0,  lane), s0);
    s0 = wmma_bf16(qa1, load_b_bf16(Kbuf, DHEAD, 0,  32, lane), s0);
    s1 = wmma_bf16(qa0, load_b_bf16(Kbuf, DHEAD, 16, 0,  lane), s1);
    s1 = wmma_bf16(qa1, load_b_bf16(Kbuf, DHEAD, 16, 32, lane), s1);

    float p0[8], p1[8], corr[8];
#pragma unroll
    for (int r = 0; r < 8; ++r) {
      int mg = mg_base + r;
      float a0 = s0[r] * 0.125f;               // 1/sqrt(64)
      float a1 = s1[r] * 0.125f;
      if (kv0 + nn > mg)      a0 = -3.0e38f;   // causal mask
      if (kv0 + 16 + nn > mg) a1 = -3.0e38f;
      float t = fmaxf(a0, a1);
#pragma unroll
      for (int msk = 1; msk < 16; msk <<= 1) t = fmaxf(t, __shfl_xor(t, msk, 32));
      float newm = fmaxf(mrow[r], t);
      float cr = __expf(mrow[r] - newm);
      float e0 = __expf(a0 - newm), e1 = __expf(a1 - newm);
      float rs = e0 + e1;
#pragma unroll
      for (int msk = 1; msk < 16; msk <<= 1) rs += __shfl_xor(rs, msk, 32);
      lrow[r] = lrow[r] * cr + rs;
      mrow[r] = newm;
      corr[r] = cr; p0[r] = e0; p1[r] = e1;
    }
#pragma unroll
    for (int r = 0; r < 8; ++r) {
      o0[r] *= corr[r]; o1[r] *= corr[r]; o2[r] *= corr[r]; o3[r] *= corr[r];
    }
    // Route P through LDS: C/D layout -> A-fragment layout
#pragma unroll
    for (int r = 0; r < 8; ++r) {
      int mm = r + ((lane >> 4) << 3);
      Plds[mm * 32 + nn]      = p0[r];
      Plds[mm * 32 + 16 + nn] = p1[r];
    }
    asm volatile("s_wait_dscnt 0" ::: "memory");
    v16bf pa;
    {
      const float* pr = &Plds[(lane & 15) * 32 + ((lane >> 4) << 3)];
#pragma unroll
      for (int i = 0; i < 8; ++i) { pa[i] = (bf16_t)pr[i]; pa[i + 8] = (bf16_t)pr[16 + i]; }
    }
    // O += P @ V  (V^T tile in LDS supplies contiguous B-fragments)
    o0 = wmma_bf16(pa, load_b_bf16(Vbuf, 32, 0,  0, lane), o0);
    o1 = wmma_bf16(pa, load_b_bf16(Vbuf, 32, 16, 0, lane), o1);
    o2 = wmma_bf16(pa, load_b_bf16(Vbuf, 32, 32, 0, lane), o2);
    o3 = wmma_bf16(pa, load_b_bf16(Vbuf, 32, 48, 0, lane), o3);
    asm volatile("s_wait_dscnt 0" ::: "memory");   // WAR on Plds vs next iter
  }
#pragma unroll
  for (int r = 0; r < 8; ++r) {
    int mm = r + ((lane >> 4) << 3);
    size_t row = (size_t)(b * S_LEN + q0 + mm);
    float inv = 1.f / lrow[r];
    attn[row * DMODEL + h * DHEAD + 0  + nn] = o0[r] * inv;
    attn[row * DMODEL + h * DHEAD + 16 + nn] = o1[r] * inv;
    attn[row * DMODEL + h * DHEAD + 32 + nn] = o2[r] * inv;
    attn[row * DMODEL + h * DHEAD + 48 + nn] = o3[r] * inv;
  }
}

// ---------------- output projection (16x128 tile per wave) ------------------
__global__ void out_proj(const float* __restrict__ attn, const bf16_t* __restrict__ wT,
                         const float* __restrict__ bias, float* __restrict__ out) {
  int lane = threadIdx.x;
  int row0 = blockIdx.x * 16;
  int col0 = blockIdx.y * 128;
  v8f acc[8] = {};
  for (int c = 0; c < DMODEL; c += 32) {
    v16bf a = load_a_f32(attn, DMODEL, row0, c, lane);
#pragma unroll
    for (int t = 0; t < 8; ++t)
      acc[t] = wmma_bf16(a, load_b_bf16(wT, DMODEL, col0 + 16 * t, c, lane), acc[t]);
  }
  int nn = lane & 15;
#pragma unroll
  for (int t = 0; t < 8; ++t) {
#pragma unroll
    for (int r = 0; r < 8; ++r) {
      int mm = r + ((lane >> 4) << 3);
      size_t row = (size_t)(row0 + mm);
      out[row * DMODEL + col0 + 16 * t + nn] = acc[t][r] + bias[col0 + 16 * t + nn];
    }
  }
}

// ---------------- host launcher ----------------------------------------------
extern "C" void kernel_launch(void* const* d_in, const int* in_sizes, int n_in,
                              void* d_out, int out_size, void* d_ws, size_t ws_size,
                              hipStream_t stream) {
  (void)in_sizes; (void)n_in; (void)out_size; (void)ws_size;
  const float* x     = (const float*)d_in[0];
  const float* w_qkv = (const float*)d_in[1];
  const float* b_qkv = (const float*)d_in[2];
  const float* w_out = (const float*)d_in[3];
  const float* b_out = (const float*)d_in[4];
  float* out = (float*)d_out;

  // workspace layout (bytes)
  char* ws = (char*)d_ws;
  bf16_t* x_bf   = (bf16_t*)(ws);                               //  8,388,608
  bf16_t* wqkvT  = (bf16_t*)(ws + 8388608);                     //  2,359,296
  bf16_t* woutT  = (bf16_t*)(ws + 10747904);                    //  2,097,152
  float*  qkv    = (float*) (ws + 12845056);                    // 18,874,368
  bf16_t* q_bf   = (bf16_t*)(ws + 31719424);                    //  8,388,608
  bf16_t* k_bf   = (bf16_t*)(ws + 40108032);                    //    524,288
  bf16_t* vT_bf  = (bf16_t*)(ws + 40632320);                    //    524,288
  float*  attn   = (float*) (ws + 41156608);                    // 16,777,216

  // 1) cast & transpose operands to bf16 (weights once, N x K for B-fragments)
  cast_f32_bf16<<<(NROWS * DMODEL + 255) / 256, 256, 0, stream>>>(x, x_bf, NROWS * DMODEL);
  transpose_cast<<<(DMODEL * DQKV + 255) / 256, 256, 0, stream>>>(w_qkv, wqkvT, DMODEL, DQKV);
  transpose_cast<<<(DMODEL * DMODEL + 255) / 256, 256, 0, stream>>>(w_out, woutT, DMODEL, DMODEL);

  // 2) qkv = x @ w_qkv + b_qkv
  qkv_gemm<<<dim3(NROWS / 16, DQKV / 128), 32, 0, stream>>>(x_bf, wqkvT, b_qkv, qkv);

  // 3) RoPE + split into q_bf / k_bf / V^T
  rope_split<<<NROWS, 256, 0, stream>>>(qkv, q_bf, k_bf, vT_bf);

  // 4) causal flash multi-query attention (TDM-staged K/V, double-buffered)
  mqa_flash<<<dim3(S_LEN / 16, NHEADS, BATCH), 32, 0, stream>>>(q_bf, k_bf, vT_bf, attn);

  // 5) out = attn @ w_out + b_out
  out_proj<<<dim3(NROWS / 16, DMODEL / 128), 32, 0, stream>>>(attn, woutT, b_out, out);
}
